// RealAgnosticAttResidualInteractionBlock_23175643530055
// MI455X (gfx1250) — compile-verified, matrix operand order checked by reading, not run on hardware
//
#include <hip/hip_runtime.h>
#include <hip/hip_bf16.h>
#include <math.h>

// ---------------------------------------------------------------------------
// CDNA5 (gfx1250) fused MACE interaction block.
// All GEMM-shaped work -> v_wmma_f32_16x16x32_bf16 (wave32, 16x16x32 tiles).
// Edge-feature gather -> global_load_async_to_lds_b128 (ASYNCcnt path).
// ---------------------------------------------------------------------------

typedef __attribute__((ext_vector_type(16))) __bf16 v16bf;
typedef __attribute__((ext_vector_type(8)))  float  v8f;

#define WMMA_BF16(a, b, c) \
  __builtin_amdgcn_wmma_f32_16x16x32_bf16(false, (a), false, (b), (short)0, (c), false, false)

__device__ __forceinline__ v8f v8f_zero() {
  v8f z;
#pragma unroll
  for (int i = 0; i < 8; ++i) z[i] = 0.f;
  return z;
}

// ---------------------------------------------------------------------------
// Prep kernels: fp32 -> bf16 conversions, weight transpose (+scale folding)
// ---------------------------------------------------------------------------

__global__ void prep_nf16(const float* __restrict__ nf, __bf16* __restrict__ out,
                          int N, int Mpad) {
  int idx = blockIdx.x * blockDim.x + threadIdx.x;
  if (idx >= Mpad * 128) return;
  int m = idx >> 7;
  out[idx] = (m < N) ? (__bf16)nf[idx] : (__bf16)0.f;
}

// W is (K x Nc) row-major fp32; Wt is (Nc x Kpad) row-major bf16 (transposed),
// zero-padded in K, with `scale` folded in.
__global__ void prep_wt(const float* __restrict__ W, __bf16* __restrict__ Wt,
                        int K, int Kpad, int Nc, float scale) {
  int idx = blockIdx.x * blockDim.x + threadIdx.x;
  if (idx >= Nc * Kpad) return;
  int n = idx / Kpad;
  int k = idx % Kpad;
  Wt[idx] = (k < K) ? (__bf16)(W[(size_t)k * Nc + n] * scale) : (__bf16)0.f;
}

// ---------------------------------------------------------------------------
// Node GEMM: C[M,N] = A[M,K](bf16) @ Bt[N,K](bf16)^T, fp32 and/or bf16 out.
// 256 threads = 8 waves; wave w owns output tile (blockIdx.x, w).
// ---------------------------------------------------------------------------

__global__ __launch_bounds__(256) void node_gemm(
    const __bf16* __restrict__ A, const __bf16* __restrict__ Bt,
    int K, int Ntiles, float* __restrict__ Cf, int ldc,
    __bf16* __restrict__ Cb, int ldcb, int Mvalid) {
  int w = threadIdx.x >> 5;
  int lane = threadIdx.x & 31;
  if (w >= Ntiles) return;  // wave-uniform
  int mbase = blockIdx.x * 16;
  int nbase = w * 16;
  int lo = lane & 15, hi = lane >> 4;

  v8f acc = v8f_zero();
  const __bf16* arow = A + (size_t)(mbase + lo) * K + hi * 16;
  const __bf16* brow = Bt + (size_t)(nbase + lo) * K + hi * 16;
  for (int kc = 0; kc < K; kc += 32) {
    v16bf a = *(const v16bf*)(arow + kc);
    v16bf b = *(const v16bf*)(brow + kc);
    acc = WMMA_BF16(a, b, acc);
  }
#pragma unroll
  for (int r = 0; r < 8; ++r) {
    int m = mbase + r + 8 * hi;
    if (m < Mvalid) {
      float v = acc[r];
      if (Cf) Cf[(size_t)m * ldc + nbase + lo] = v;
      if (Cb) Cb[(size_t)m * ldcb + nbase + lo] = (__bf16)v;
    }
  }
}

// ---------------------------------------------------------------------------
// Fused edge pipeline: async-gather aug -> 4-layer MLP (WMMA) -> outer product
// with x_up[sender] and Y -> atomic segment-sum into message[N,2048].
// Block = 256 thr = 8 waves; each wave owns 16 edges and an 8KB LDS slab.
// ---------------------------------------------------------------------------

// One 256-wide silu layer on a wave's 16-edge activation slab (stride 256).
__device__ __forceinline__ void mlp_layer256(__bf16* smw, const __bf16* __restrict__ Wt,
                                             int K, int lane) {
  int lo = lane & 15, hi = lane >> 4;
  v8f acc[16];
#pragma unroll
  for (int nt = 0; nt < 16; ++nt) acc[nt] = v8f_zero();
  for (int kc = 0; kc < K; kc += 32) {
    v16bf a = *(const v16bf*)&smw[lo * 256 + kc + hi * 16];
#pragma unroll
    for (int nt = 0; nt < 16; ++nt) {
      v16bf b = *(const v16bf*)&Wt[(size_t)(nt * 16 + lo) * K + kc + hi * 16];
      acc[nt] = WMMA_BF16(a, b, acc[nt]);
    }
  }
  // silu + writeback (all reads above are program-order before these DS writes)
#pragma unroll
  for (int nt = 0; nt < 16; ++nt) {
#pragma unroll
    for (int r = 0; r < 8; ++r) {
      float v = acc[nt][r];
      v = v / (1.f + __expf(-v));
      smw[(r + 8 * hi) * 256 + nt * 16 + lo] = (__bf16)v;
    }
  }
}

__global__ __launch_bounds__(256, 1) void edge_mlp_scatter(
    const float* __restrict__ edge_feats, const float* __restrict__ edge_attrs,
    const int* __restrict__ sender, const int* __restrict__ receiver,
    const __bf16* __restrict__ xd, const float* __restrict__ xup,
    const __bf16* __restrict__ Wt0, const __bf16* __restrict__ Wt1,
    const __bf16* __restrict__ Wt2, const __bf16* __restrict__ Wt3,
    float* __restrict__ message, int E) {
  __shared__ __bf16 sm[8 * 16 * 256];  // 64 KB: one 16x256 slab per wave (LDS off 0)
  int w = threadIdx.x >> 5;
  int lane = threadIdx.x & 31;
  int lo = lane & 15, hi = lane >> 4;
  __bf16* smw = &sm[w * 16 * 256];
  int ebase = (blockIdx.x * 8 + w) * 16;
  unsigned slab = (unsigned)w * 16u * 256u * 2u;  // LDS byte offset of this slab

  __builtin_prefetch(Wt0, 0, 3);
  __builtin_prefetch(Wt1, 0, 3);
  __builtin_prefetch(Wt2, 0, 3);
  __builtin_prefetch(Wt3, 0, 3);

  // ---- async-gather x_down rows straight into LDS (bypasses VGPRs) --------
  // aug row layout (bf16 cols): [0..8) edge_feats  [8..72) xd[sender]
  //                             [72..136) xd[receiver]  [136..160) zero pad
  // Each (row, src) segment is 128B = 8 lanes x 16B; 32 lanes = 4 segs/inst.
  {
    int seg = lane >> 3;        // 0..3
    int sub = lane & 7;         // 0..7 -> 16B chunk within a 128B row
#pragma unroll
    for (int it = 0; it < 8; ++it) {
      int segidx = it * 4 + seg;       // 0..31
      int row = segidx >> 1;           // 0..15
      int src = segidx & 1;            // 0 = sender, 1 = receiver
      int e = ebase + row;
      int ee = (e < E) ? e : (E - 1);  // clamp (pad rows are discarded later)
      int node = src ? receiver[ee] : sender[ee];
      const __bf16* gptr = xd + (size_t)node * 64 + sub * 8;
      unsigned ldsoff = slab + (unsigned)row * 512u + 16u + (unsigned)src * 128u
                        + (unsigned)sub * 16u;
      asm volatile("global_load_async_to_lds_b128 %0, %1, off"
                   :: "v"(ldsoff), "v"(gptr) : "memory");
    }
  }

  // ---- VALU-staged part: bessel feats (fp32->bf16) + K zero padding -------
  // cols 0..8 and 136..160 -> 32 cols per row, 16 rows = 512 elems / wave.
  for (int idx = lane; idx < 16 * 32; idx += 32) {
    int row = idx >> 5;
    int j = idx & 31;
    int col = (j < 8) ? j : (128 + j);  // 0..7 or 136..159
    int e = ebase + row;
    __bf16 bv = (__bf16)0.f;
    if (j < 8 && e < E) bv = (__bf16)edge_feats[(size_t)e * 8 + col];
    smw[row * 256 + col] = bv;
  }

  asm volatile("s_wait_asynccnt 0" ::: "memory");

  // ---- 3 hidden silu layers (K = 160, 256, 256; N = 256) ----
  mlp_layer256(smw, Wt0, 160, lane);
  mlp_layer256(smw, Wt1, 256, lane);
  mlp_layer256(smw, Wt2, 256, lane);

  // ---- final layer (K=256, N=512) consumed tile-by-tile into atomic scatter ----
  int snd[8], rcv[8];
#pragma unroll
  for (int r = 0; r < 8; ++r) {
    int e = ebase + 8 * hi + r;
    bool ok = (e < E);
    snd[r] = ok ? sender[e] : 0;
    rcv[r] = ok ? receiver[e] : 0;
  }

  for (int nt = 0; nt < 32; ++nt) {
    v8f acc = v8f_zero();
    const __bf16* brow = Wt3 + (size_t)(nt * 16 + lo) * 256 + hi * 16;
    for (int kc = 0; kc < 256; kc += 32) {
      v16bf a = *(const v16bf*)&smw[lo * 256 + kc + hi * 16];
      v16bf b = *(const v16bf*)(brow + kc);
      acc = WMMA_BF16(a, b, acc);
    }
    int li = nt >> 3;          // which L (0..3): tp_w reshaped (E,4,128)
    int cbase = (nt & 7) * 16; // channel base within 128
    int dim = 2 * li + 1;
    int yoff = li * li;        // OFFS[li] = li^2
    int moff = 128 * li * li;  // LOFF[li] = 128*li^2
    float cgl = rsqrtf((float)(2 * li + 1));
    int c = cbase + lo;
#pragma unroll
    for (int r = 0; r < 8; ++r) {
      int e = ebase + 8 * hi + r;
      if (e < E) {
        float v0 = acc[r] * xup[(size_t)snd[r] * 128 + c];
        float* mrow = message + (size_t)rcv[r] * 2048 + moff + c * dim;
        const float* yrow = edge_attrs + (size_t)e * 16 + yoff;
        for (int d = 0; d < dim; ++d) {
          atomicAdd(&mrow[d], v0 * cgl * yrow[d]);
        }
      }
    }
  }
}

// ---------------------------------------------------------------------------
// Output einsum: for each (l, d): out[:, :, li^2+d] slice = message-slice @
// W_out[l] (pre-transposed bf16, 0.1 folded).  A is strided fp32 -> bf16.
// grid = (Mpad/16, 16 (l,d)-pairs); 8 waves = 8 k-column tiles of 128.
// ---------------------------------------------------------------------------

__global__ __launch_bounds__(256) void out_gemm(
    const float* __restrict__ message, const __bf16* __restrict__ WoutT,
    float* __restrict__ out, int Mvalid) {
  int w = threadIdx.x >> 5;
  int lane = threadIdx.x & 31;
  int lo = lane & 15, hi = lane >> 4;
  int mbase = blockIdx.x * 16;
  int pair = blockIdx.y;  // 0..15 -> (li, dofs)
  int li = 0;
  while ((li + 1) * (li + 1) <= pair) ++li;
  int dofs = pair - li * li;
  int dim = 2 * li + 1;
  int loff = 128 * li * li;

  const float* arow = message + (size_t)(mbase + lo) * 2048 + loff + dofs;
  const __bf16* brow = WoutT + (size_t)li * 128 * 128 + (size_t)(w * 16 + lo) * 128 + hi * 16;

  v8f acc = v8f_zero();
  for (int kc = 0; kc < 128; kc += 32) {
    v16bf a;
#pragma unroll
    for (int i = 0; i < 16; ++i) {
      int k = kc + hi * 16 + i;
      a[i] = (__bf16)arow[(size_t)k * dim];
    }
    v16bf b = *(const v16bf*)(brow + kc);
    acc = WMMA_BF16(a, b, acc);
  }
#pragma unroll
  for (int r = 0; r < 8; ++r) {
    int m = mbase + r + 8 * hi;
    if (m < Mvalid) {
      // out shape (N, 128, 16): (m, k, li^2 + dofs)
      out[(size_t)m * 2048 + (size_t)(w * 16 + lo) * 16 + li * li + dofs] = acc[r];
    }
  }
}

// ---------------------------------------------------------------------------
// Host launch
// ---------------------------------------------------------------------------

extern "C" void kernel_launch(void* const* d_in, const int* in_sizes, int n_in,
                              void* d_out, int out_size, void* d_ws, size_t ws_size,
                              hipStream_t stream) {
  (void)n_in; (void)out_size; (void)ws_size;
  // setup_inputs order:
  // 0 node_attrs (unused), 1 node_feats, 2 edge_attrs, 3 edge_feats,
  // 4 edge_index, 5 W_up, 6 W_down, 7 W_skip, 8..11 W_fc0..3, 12 W_out
  const float* node_feats = (const float*)d_in[1];
  const float* edge_attrs = (const float*)d_in[2];
  const float* edge_feats = (const float*)d_in[3];
  const int*   edge_index = (const int*)d_in[4];
  const float* W_up   = (const float*)d_in[5];
  const float* W_down = (const float*)d_in[6];
  const float* W_skip = (const float*)d_in[7];
  const float* W_fc0  = (const float*)d_in[8];
  const float* W_fc1  = (const float*)d_in[9];
  const float* W_fc2  = (const float*)d_in[10];
  const float* W_fc3  = (const float*)d_in[11];
  const float* W_out  = (const float*)d_in[12];

  const int N = in_sizes[1] / 128;
  const int E = in_sizes[4] / 2;
  const int Mpad = ((N + 15) / 16) * 16;
  const int Mtiles = Mpad / 16;

  const int* sender   = edge_index;
  const int* receiver = edge_index + E;

  // --- workspace layout (256B-aligned slices) ---
  char* base = (char*)d_ws;
  size_t off = 0;
  auto take = [&](size_t bytes) -> char* {
    char* p = base + off;
    off = (off + bytes + 255) & ~(size_t)255;
    return p;
  };
  __bf16* nf16    = (__bf16*)take((size_t)Mpad * 128 * 2);
  __bf16* WskipT  = (__bf16*)take(128 * 128 * 2);
  __bf16* WupT    = (__bf16*)take(128 * 128 * 2);
  __bf16* WdownT  = (__bf16*)take(64 * 128 * 2);
  __bf16* Wfc0T   = (__bf16*)take(256 * 160 * 2);
  __bf16* Wfc1T   = (__bf16*)take(256 * 256 * 2);
  __bf16* Wfc2T   = (__bf16*)take(256 * 256 * 2);
  __bf16* Wfc3T   = (__bf16*)take(512 * 256 * 2);
  __bf16* WoutT   = (__bf16*)take(4 * 128 * 128 * 2);
  float*  xup     = (float*)take((size_t)Mpad * 128 * 4);
  __bf16* xdown   = (__bf16*)take((size_t)Mpad * 64 * 2);
  float*  message = (float*)take((size_t)Mpad * 2048 * 4);

  hipMemsetAsync(message, 0, (size_t)Mpad * 2048 * 4, stream);

  // --- prep ---
  {
    int n = Mpad * 128;
    prep_nf16<<<(n + 255) / 256, 256, 0, stream>>>(node_feats, nf16, N, Mpad);
  }
  auto launch_wt = [&](const float* W, __bf16* Wt, int K, int Kpad, int Nc, float scale) {
    int n = Nc * Kpad;
    prep_wt<<<(n + 255) / 256, 256, 0, stream>>>(W, Wt, K, Kpad, Nc, scale);
  };
  launch_wt(W_skip, WskipT, 128, 128, 128, 1.0f);
  launch_wt(W_up,   WupT,   128, 128, 128, 1.0f);
  launch_wt(W_down, WdownT, 128, 128, 64,  1.0f);
  launch_wt(W_fc0,  Wfc0T,  136, 160, 256, 1.0f / sqrtf(136.f));
  launch_wt(W_fc1,  Wfc1T,  256, 256, 256, 1.0f / 16.f);
  launch_wt(W_fc2,  Wfc2T,  256, 256, 256, 1.0f / 16.f);
  launch_wt(W_fc3,  Wfc3T,  256, 256, 512, 1.0f / 16.f);
  for (int l = 0; l < 4; ++l) {  // fold 1/AVG_NUM_NEIGHBORS here
    launch_wt(W_out + (size_t)l * 128 * 128, WoutT + (size_t)l * 128 * 128,
              128, 128, 128, 0.1f);
  }

  // --- node GEMMs ---
  float* sc_out = (float*)d_out + (size_t)N * 2048;  // second tuple element
  node_gemm<<<Mtiles, 256, 0, stream>>>(nf16, WskipT, 128, 8, sc_out, 128,
                                        (__bf16*)nullptr, 0, N);
  node_gemm<<<Mtiles, 256, 0, stream>>>(nf16, WupT, 128, 8, xup, 128,
                                        (__bf16*)nullptr, 0, N);
  node_gemm<<<Mtiles, 256, 0, stream>>>(nf16, WdownT, 128, 4, (float*)nullptr, 0,
                                        xdown, 64, N);

  // --- fused edge MLP + tensor product + segment-sum scatter ---
  {
    int blocks = (E + 127) / 128;
    edge_mlp_scatter<<<blocks, 256, 0, stream>>>(
        edge_feats, edge_attrs, sender, receiver, xdown, xup,
        Wfc0T, Wfc1T, Wfc2T, Wfc3T, message, E);
  }

  // --- output einsum ---
  {
    dim3 grid(Mtiles, 16);
    out_gemm<<<grid, 256, 0, stream>>>(message, WoutT, (float*)d_out, N);
  }
}
